// GraphEncoderGAT_6536940224755
// MI455X (gfx1250) — matrix operand order repaired and checked by reading, not compiled, not run on hardware
//
#include <hip/hip_runtime.h>
#include <hip/hip_fp16.h>

typedef __attribute__((ext_vector_type(16))) _Float16 v16h;
typedef __attribute__((ext_vector_type(8)))  _Float16 v8h;
typedef __attribute__((ext_vector_type(8)))  float    v8f;

#define HEADS 4
#define CH    32
#define OUTF  128   // HEADS*CH
#define TB    256

// ---------------- conversion / packing ----------------
__global__ void k_f32_to_f16(const float* __restrict__ src, _Float16* __restrict__ dst, int n) {
  int i = blockIdx.x * blockDim.x + threadIdx.x;
  if (i < n) dst[i] = (_Float16)src[i];
}

// Pack W [K x 128] (row-major, f32) into CDNA5 WMMA B-operand order (f16):
// Bp[((c*Ksteps+ks)*32+lane)*16 + t] = W[ks*32 + (lane>=16?8:0) + (t&7) + (t>=8?16:0)][c*16 + (lane&15)]
// (16-bit B 32x16: lanes 0-15 = cols N, K = {0..7, 16..23}; lanes 16-31 hold K = {8..15, 24..31})
__global__ void k_pack_B(const float* __restrict__ W, _Float16* __restrict__ Bp, int Ksteps) {
  int idx = blockIdx.x * blockDim.x + threadIdx.x;
  int total = 8 * Ksteps * 32 * 16;
  if (idx >= total) return;
  int t    = idx & 15;
  int lane = (idx >> 4) & 31;
  int rest = idx >> 9;            // c*Ksteps + ks
  int ks   = rest % Ksteps;
  int c    = rest / Ksteps;
  int k = ks * 32 + ((lane >> 4) << 3) + (t & 7) + ((t >> 3) << 4);
  int n = c * 16 + (lane & 15);
  Bp[idx] = (_Float16)W[k * 128 + n];
}

// ---------------- WMMA GEMM: H[nrows x 128] = Xh[nrows x (32*Ksteps)] * W ----------------
__global__ void k_gemm_wmma(const _Float16* __restrict__ Xh, const _Float16* __restrict__ Bp,
                            float* __restrict__ H, int nrows, int Ksteps) {
  const int Kin  = Ksteps * 32;
  const int lane = threadIdx.x & 31;
  const int wv   = threadIdx.x >> 5;       // 0..7 : 16-col tile owned by this wave
  const int row0 = blockIdx.x * 16;
  int rr = row0 + (lane & 15);
  int r  = rr < nrows ? rr : nrows - 1;    // clamp loads; stores predicated after WMMA

  v8f acc = {};
  for (int ks = 0; ks < Ksteps; ++ks) {
    // A (16x32 f16): lane<16 -> row=lane, K = {k0..k0+7, k0+16..k0+23}
    //                lane>=16 -> row=lane-16, K = {k0+8..15, k0+24..31}
    int kb = ks * 32 + ((lane >> 4) << 3);
    const _Float16* ap = Xh + (size_t)r * Kin + kb;
    v8h a0 = *(const v8h*)(ap);
    v8h a1 = *(const v8h*)(ap + 16);
    v16h A;
#pragma unroll
    for (int j = 0; j < 8; ++j) { A[j] = a0[j]; A[j + 8] = a1[j]; }
    v16h B = *(const v16h*)(Bp + (size_t)((wv * Ksteps + ks) * 32 + lane) * 16);
    acc = __builtin_amdgcn_wmma_f32_16x16x32_f16(false, A, false, B, (short)0, acc, false, false);
  }
  // C/D layout: lane holds col N = lane&15; rows M = (lane>=16?8:0)+i in acc[i]
  int col = wv * 16 + (lane & 15);
  int mb  = (lane >> 4) * 8;
#pragma unroll
  for (int i = 0; i < 8; ++i) {
    int rowi = row0 + mb + i;
    if (rowi < nrows) H[(size_t)rowi * OUTF + col] = acc[i];
  }
}

// ---------------- attention coefficients ----------------
__global__ void k_attn(const float* __restrict__ H, const float* __restrict__ a_src,
                       const float* __restrict__ a_dst, float* __restrict__ als,
                       float* __restrict__ ald, int N) {
  int idx = blockIdx.x * blockDim.x + threadIdx.x;
  if (idx >= N * HEADS) return;
  int n = idx >> 2, hh = idx & 3;
  const float* hp = H + (size_t)n * OUTF + hh * CH;
  const float* as = a_src + hh * CH;
  const float* ad = a_dst + hh * CH;
  float s = 0.f, d = 0.f;
#pragma unroll
  for (int c = 0; c < CH; ++c) { float v = hp[c]; s += v * as[c]; d += v * ad[c]; }
  als[idx] = s; ald[idx] = d;
}

__global__ void k_fill(float* __restrict__ p, float v, size_t n) {
  size_t i = (size_t)blockIdx.x * blockDim.x + threadIdx.x;
  if (i < n) p[i] = v;
}

// ---------------- edge kernels ----------------
__device__ __forceinline__ void edge_sd(int e, const int* __restrict__ ei, int E, int& s, int& d) {
  if (e < E) { s = ei[e]; d = ei[E + e]; } else { s = d = e - E; }  // self-loops appended
}

__device__ __forceinline__ void atomicMaxF(float* addr, float val) {
  unsigned int* ua = (unsigned int*)addr;
  unsigned int old = *ua;
  while (__uint_as_float(old) < val) {
    unsigned int assumed = old;
    old = atomicCAS(ua, assumed, __float_as_uint(val));
    if (old == assumed) break;
  }
}

__global__ void k_edge_logits(const float* __restrict__ als, const float* __restrict__ ald,
                              const int* __restrict__ ei, int E, int N,
                              float* __restrict__ eb, float* __restrict__ m) {
  int e = blockIdx.x * blockDim.x + threadIdx.x;
  if (e >= E + N) return;
  int s, d; edge_sd(e, ei, E, s, d);
#pragma unroll
  for (int hh = 0; hh < HEADS; ++hh) {
    float v = als[s * HEADS + hh] + ald[d * HEADS + hh];
    v = v >= 0.f ? v : 0.2f * v;              // leaky_relu(0.2)
    eb[(size_t)e * HEADS + hh] = v;
    atomicMaxF(&m[d * HEADS + hh], v);        // segment max
  }
}

__global__ void k_edge_exp(const int* __restrict__ ei, int E, int N,
                           const float* __restrict__ m, float* __restrict__ eb,
                           float* __restrict__ denom) {
  int e = blockIdx.x * blockDim.x + threadIdx.x;
  if (e >= E + N) return;
  int s, d; edge_sd(e, ei, E, s, d);
#pragma unroll
  for (int hh = 0; hh < HEADS; ++hh) {
    float ex = __expf(eb[(size_t)e * HEADS + hh] - m[d * HEADS + hh]);
    eb[(size_t)e * HEADS + hh] = ex;
    atomicAdd(&denom[d * HEADS + hh], ex);    // segment sum
  }
}

// 1 thread per (edge, out-feature); coalesced gather of H[src], coalesced atomics on agg[dst]
__global__ void k_edge_agg(const float* __restrict__ H, const float* __restrict__ eb,
                           const float* __restrict__ denom, const int* __restrict__ ei,
                           int E, int N, float* __restrict__ agg) {
  long long idx = (long long)blockIdx.x * blockDim.x + threadIdx.x;
  long long total = (long long)(E + N) * OUTF;
  if (idx >= total) return;
  int e = (int)(idx >> 7);
  int t = (int)(idx & (OUTF - 1));
  int hh = t >> 5;
  int s, d; edge_sd(e, ei, E, s, d);
  float alpha = eb[(size_t)e * HEADS + hh] / (denom[d * HEADS + hh] + 1e-16f);
  atomicAdd(&agg[(size_t)d * OUTF + t], H[(size_t)s * OUTF + t] * alpha);
}

// mean over heads + bias + ELU; optional f32 and f16 outputs
__global__ void k_finalize(const float* __restrict__ agg, const float* __restrict__ b,
                           float* __restrict__ out32, _Float16* __restrict__ out16, int N) {
  int idx = blockIdx.x * blockDim.x + threadIdx.x;
  if (idx >= N * CH) return;
  int n = idx >> 5, c = idx & (CH - 1);
  float v = 0.f;
#pragma unroll
  for (int hh = 0; hh < HEADS; ++hh) v += agg[(size_t)n * OUTF + hh * CH + c];
  v = v * 0.25f + b[c];
  v = v > 0.f ? v : (__expf(v) - 1.f);        // ELU
  if (out32) out32[idx] = v;
  if (out16) out16[idx] = (_Float16)v;
}

// ---------------- global mean pool ----------------
__global__ void k_pool_acc(const float* __restrict__ x, const int* __restrict__ batch,
                           float* __restrict__ sums, float* __restrict__ cnt, int N) {
  int idx = blockIdx.x * blockDim.x + threadIdx.x;
  if (idx >= N * CH) return;
  int n = idx >> 5, c = idx & (CH - 1);
  int g = batch[n];
  atomicAdd(&sums[g * CH + c], x[idx]);
  if (c == 0) atomicAdd(&cnt[g], 1.0f);
}

__global__ void k_pool_div(const float* __restrict__ sums, const float* __restrict__ cnt,
                           float* __restrict__ out, int G) {
  int idx = blockIdx.x * blockDim.x + threadIdx.x;
  if (idx >= G * CH) return;
  out[idx] = sums[idx] / fmaxf(cnt[idx >> 5], 1.0f);
}

// ---------------- driver ----------------
extern "C" void kernel_launch(void* const* d_in, const int* in_sizes, int n_in,
                              void* d_out, int out_size, void* d_ws, size_t ws_size,
                              hipStream_t stream) {
  const float* x      = (const float*)d_in[0];
  const float* W1     = (const float*)d_in[1];
  const float* asrc1  = (const float*)d_in[2];
  const float* adst1  = (const float*)d_in[3];
  const float* b1     = (const float*)d_in[4];
  const float* W2     = (const float*)d_in[5];
  const float* asrc2  = (const float*)d_in[6];
  const float* adst2  = (const float*)d_in[7];
  const float* b2     = (const float*)d_in[8];
  const int*   ei     = (const int*)d_in[9];
  const int*   batch  = (const int*)d_in[10];

  const int N = in_sizes[0] / 128;     // 50000
  const int E = in_sizes[9] / 2;       // 800000
  const int G = out_size / CH;         // 64
  const int EN = E + N;

  // carve workspace
  char* ws = (char*)d_ws; size_t off = 0;
  auto carve = [&](size_t bytes) -> void* {
    void* p = ws + off; off = (off + bytes + 255) & ~(size_t)255; return p;
  };
  _Float16* xh   = (_Float16*)carve((size_t)N * 128 * 2);
  _Float16* w1p  = (_Float16*)carve(8 * 4 * 32 * 16 * 2);
  _Float16* w2p  = (_Float16*)carve(8 * 1 * 32 * 16 * 2);
  float*    hbuf = (float*)carve((size_t)N * OUTF * 4);
  float*    als  = (float*)carve((size_t)N * HEADS * 4);
  float*    ald  = (float*)carve((size_t)N * HEADS * 4);
  float*    mbuf = (float*)carve((size_t)N * HEADS * 4);
  float*    den  = (float*)carve((size_t)N * HEADS * 4);
  float*    eb   = (float*)carve((size_t)EN * HEADS * 4);
  float*    agg  = (float*)carve((size_t)N * OUTF * 4);
  _Float16* x2h  = (_Float16*)carve((size_t)N * CH * 2);
  float*    fin2 = (float*)carve((size_t)N * CH * 4);
  float*    sums = (float*)carve((size_t)G * CH * 4);
  float*    cnt  = (float*)carve((size_t)G * 4);

  auto blocks = [](long long n) { return (unsigned)((n + TB - 1) / TB); };
  const unsigned gb16 = (unsigned)((N + 15) / 16);

  // prep: f16 input + packed weights
  k_f32_to_f16<<<blocks((long long)N * 128), TB, 0, stream>>>(x, xh, N * 128);
  k_pack_B<<<blocks(8 * 4 * 32 * 16), TB, 0, stream>>>(W1, w1p, 4);
  k_pack_B<<<blocks(8 * 1 * 32 * 16), TB, 0, stream>>>(W2, w2p, 1);

  for (int layer = 0; layer < 2; ++layer) {
    // GEMM (WMMA)
    if (layer == 0)
      k_gemm_wmma<<<gb16, TB, 0, stream>>>(xh, w1p, hbuf, N, 4);
    else
      k_gemm_wmma<<<gb16, TB, 0, stream>>>(x2h, w2p, hbuf, N, 1);

    const float* as = layer == 0 ? asrc1 : asrc2;
    const float* ad = layer == 0 ? adst1 : adst2;
    const float* bb = layer == 0 ? b1 : b2;

    k_attn<<<blocks((long long)N * HEADS), TB, 0, stream>>>(hbuf, as, ad, als, ald, N);

    k_fill<<<blocks((long long)N * HEADS), TB, 0, stream>>>(mbuf, -__builtin_huge_valf(), (size_t)N * HEADS);
    k_fill<<<blocks((long long)N * HEADS), TB, 0, stream>>>(den, 0.f, (size_t)N * HEADS);
    k_fill<<<blocks((long long)N * OUTF), TB, 0, stream>>>(agg, 0.f, (size_t)N * OUTF);

    k_edge_logits<<<blocks(EN), TB, 0, stream>>>(als, ald, ei, E, N, eb, mbuf);
    k_edge_exp<<<blocks(EN), TB, 0, stream>>>(ei, E, N, mbuf, eb, den);
    k_edge_agg<<<blocks((long long)EN * OUTF), TB, 0, stream>>>(hbuf, eb, den, ei, E, N, agg);

    if (layer == 0)
      k_finalize<<<blocks((long long)N * CH), TB, 0, stream>>>(agg, bb, (float*)nullptr, x2h, N);
    else
      k_finalize<<<blocks((long long)N * CH), TB, 0, stream>>>(agg, bb, fin2, (_Float16*)nullptr, N);
  }

  // global mean pool
  k_fill<<<blocks((long long)G * CH), TB, 0, stream>>>(sums, 0.f, (size_t)G * CH);
  k_fill<<<blocks(G), TB, 0, stream>>>(cnt, 0.f, (size_t)G);
  k_pool_acc<<<blocks((long long)N * CH), TB, 0, stream>>>(fin2, batch, sums, cnt, N);
  k_pool_div<<<blocks((long long)G * CH), TB, 0, stream>>>(sums, cnt, (float*)d_out, G);
}
